// GraphTransformerBlock_6433861009675
// MI455X (gfx1250) — compile-verified, matrix operand order checked by reading, not run on hardware
//
#include <hip/hip_runtime.h>
#include <hip/hip_bf16.h>
#include <math.h>

// ---------------------------------------------------------------------------
// Problem constants
// ---------------------------------------------------------------------------
constexpr int N   = 50000;
constexpr int E   = 512000;
constexpr int E2  = E / 2;       // 256000
constexpr int DIN = 128;
constexpr int HC  = 128;         // H*C
constexpr int C   = 32;
constexpr float SCALE = 0.17677669529663687f;   // 1/sqrt(32)

typedef __attribute__((ext_vector_type(16))) __bf16 v16bf;
typedef __attribute__((ext_vector_type(8)))  float  v8f;

#define WMMA_BF16(acc, a, b) \
  acc = __builtin_amdgcn_wmma_f32_16x16x32_bf16(false, (a), false, (b), (short)0, (acc), false, false)

// ---------------------------------------------------------------------------
// Workspace layout (element offsets; floats unless noted)
// ---------------------------------------------------------------------------
constexpr size_t F_Q      = 0;
constexpr size_t F_K      = F_Q      + (size_t)N * HC;
constexpr size_t F_VCON   = F_K      + (size_t)N * HC;
constexpr size_t F_ATTR   = F_VCON   + (size_t)N * HC;
constexpr size_t F_XRAW   = F_ATTR   + (size_t)N * C;
constexpr size_t F_L1     = F_XRAW   + (size_t)N * C;
constexpr size_t F_L2     = F_L1     + (size_t)E * 4;
constexpr size_t F_S1     = F_L2     + (size_t)E * 4;
constexpr size_t F_S2     = F_S1     + (size_t)N * 4;
constexpr size_t F_ACC1   = F_S2     + (size_t)N * 4;
constexpr size_t F_ACC2   = F_ACC1   + (size_t)N * HC;
constexpr size_t F_XH     = F_ACC2   + (size_t)N * HC;
constexpr size_t F_DEGB   = F_XH     + (size_t)E2 * C;
constexpr size_t F_O1ACC  = F_DEGB   + (size_t)N;
constexpr size_t F_O1     = F_O1ACC  + (size_t)N * C;
constexpr size_t F_SUMS   = F_O1     + (size_t)N * C;      // 64: sum[32], sumsq[32]
constexpr size_t F_M3     = F_SUMS   + 64;                 // 128*128 f32 composite
constexpr size_t F_BE3CON = F_M3     + 128 * 128;
constexpr size_t F_M1     = F_BE3CON + 128;                // unsigned (ordered-encoded)
constexpr size_t F_M2     = F_M1     + (size_t)N * 4;
constexpr size_t F_END    = F_M2     + (size_t)N * 4;
constexpr size_t PACK_BYTE = ((F_END * 4 + 1023) / 1024) * 1024;
// packed bf16 B-fragment region (v16bf units, 32 lanes per fragment)
constexpr size_t P_Q   = 0;            // 32 frags
constexpr size_t P_K   = P_Q   + 32 * 32;
constexpr size_t P_V   = P_K   + 32 * 32;
constexpr size_t P_E2  = P_V   + 32 * 32;
constexpr size_t P_M3  = P_E2  + 32 * 32;
constexpr size_t P_CON = P_M3  + 32 * 32;  // 8 frags
constexpr size_t P_H1  = P_CON + 8 * 32;
constexpr size_t P_H2  = P_H1  + 8 * 32;

constexpr unsigned ENC_NEG_INF = 0x007FFFFFu;   // enc_ord(-inf)

// ---------------------------------------------------------------------------
// Helpers
// ---------------------------------------------------------------------------
__device__ inline unsigned enc_ord(float f) {
  unsigned b = __float_as_uint(f);
  return (b & 0x80000000u) ? ~b : (b | 0x80000000u);
}
__device__ inline float dec_ord(unsigned k) {
  unsigned b = (k & 0x80000000u) ? (k ^ 0x80000000u) : ~k;
  return __uint_as_float(b);
}

// A fragment for v_wmma_f32_16x16x32_bf16 from a row-major f32 tile.
// p0 points at (row0, kblock); lane<16 -> M=lane, K in {0..7,16..23};
// lane>=16 -> M=lane-16, K in {8..15,24..31}.
__device__ inline v16bf load_fragA(const float* p0, int ld, int lane) {
  int m = lane & 15;
  int kbase = (lane < 16) ? 0 : 8;
  const float* p = p0 + (size_t)m * ld + kbase;
  v16bf r;
#pragma unroll
  for (int j = 0; j < 8; ++j) r[j] = (__bf16)p[j];
#pragma unroll
  for (int j = 0; j < 8; ++j) r[8 + j] = (__bf16)p[16 + j];
  return r;
}

// D/C layout: element r of v8f at lane -> row r + (lane<16?0:8), col = lane&15
__device__ inline void store_D(float* base, int ld, int lane, v8f d) {
  int col = lane & 15;
  int r0 = (lane < 16) ? 0 : 8;
#pragma unroll
  for (int r = 0; r < 8; ++r) base[(size_t)(r0 + r) * ld + col] = d[r];
}

__device__ inline v8f bias_bcast(const float* b, int nb, int lane) {
  float bv = b[nb * 16 + (lane & 15)];
  v8f a;
#pragma unroll
  for (int r = 0; r < 8; ++r) a[r] = bv;
  return a;
}
__device__ inline v8f zero8() {
  v8f a;
#pragma unroll
  for (int r = 0; r < 8; ++r) a[r] = 0.0f;
  return a;
}

// ---------------------------------------------------------------------------
// Prep: pack a row-major f32 weight [Kdim x NC] into per-lane WMMA B-fragments
// frag = kb*(NC/16)+nb ; B[K][n]: lane holds n=nb*16+(lane&15),
// K = kb*32 + kbase + {0..7, 16..23} (kbase = 0 or 8 by lane half).
// ---------------------------------------------------------------------------
__global__ void pack_b_kernel(const float* __restrict__ W, v16bf* __restrict__ out, int NC) {
  int frag = blockIdx.x;
  int lane = threadIdx.x;
  int nfragN = NC / 16;
  int kb = frag / nfragN, nb = frag % nfragN;
  int n = nb * 16 + (lane & 15);
  int kbase = kb * 32 + ((lane < 16) ? 0 : 8);
  v16bf r;
#pragma unroll
  for (int j = 0; j < 8; ++j) r[j] = (__bf16)W[(size_t)(kbase + j) * NC + n];
#pragma unroll
  for (int j = 0; j < 8; ++j) r[8 + j] = (__bf16)W[(size_t)(kbase + 16 + j) * NC + n];
  out[(size_t)frag * 32 + lane] = r;
}

// M3[d][h*32+c2] = sum_c We3[d][h*32+c] * Wcon[h*32+c][c2]; be3con likewise.
__global__ void build_m3_kernel(const float* __restrict__ We3, const float* __restrict__ Wcon,
                                const float* __restrict__ be3, float* __restrict__ M3,
                                float* __restrict__ be3con) {
  int idx = blockIdx.x * blockDim.x + threadIdx.x;
  if (idx < 128 * 128) {
    int d = idx >> 7, col = idx & 127;
    int h = col >> 5, c2 = col & 31;
    float s = 0.0f;
    for (int c = 0; c < 32; ++c) s += We3[d * 128 + h * 32 + c] * Wcon[(h * 32 + c) * 32 + c2];
    M3[idx] = s;
  }
  if (idx < 128) {
    int h = idx >> 5, c2 = idx & 31;
    float s = 0.0f;
    for (int c = 0; c < 32; ++c) s += be3[h * 32 + c] * Wcon[(h * 32 + c) * 32 + c2];
    be3con[idx] = s;
  }
}

__global__ void init_kernel(unsigned* __restrict__ m1, unsigned* __restrict__ m2,
                            float* __restrict__ s1, float* __restrict__ s2,
                            float* __restrict__ acc1, float* __restrict__ acc2,
                            float* __restrict__ sums, float* __restrict__ degb,
                            float* __restrict__ o1acc) {
  size_t idx = (size_t)blockIdx.x * blockDim.x + threadIdx.x;
  if (idx < (size_t)N * 4) { m1[idx] = ENC_NEG_INF; m2[idx] = ENC_NEG_INF; s1[idx] = 0.f; s2[idx] = 0.f; }
  if (idx < (size_t)N * HC) { acc1[idx] = 0.f; acc2[idx] = 0.f; }
  if (idx < 64) sums[idx] = 0.f;
  if (idx < (size_t)N) degb[idx] = 0.f;
  if (idx < (size_t)N * C) o1acc[idx] = 0.f;
}

// ---------------------------------------------------------------------------
// Node GEMMs: q,k (stored), v -> folded through Wcon per head -> vcon, attr=x@Wh2
// one wave (32 lanes) per 16-row tile
// ---------------------------------------------------------------------------
__global__ __launch_bounds__(32) void node_gemm_kernel(
    const float* __restrict__ x,
    const float* __restrict__ bq, const float* __restrict__ bk, const float* __restrict__ bv,
    const v16bf* __restrict__ pQ, const v16bf* __restrict__ pK, const v16bf* __restrict__ pV,
    const v16bf* __restrict__ pCon, const v16bf* __restrict__ pH2,
    float* __restrict__ q, float* __restrict__ k, float* __restrict__ vcon,
    float* __restrict__ attr) {
  __shared__ float vtile[16][132];
  int lane = threadIdx.x;
  int row0 = blockIdx.x * 16;

  v16bf A[4];
#pragma unroll
  for (int kb = 0; kb < 4; ++kb)
    A[kb] = load_fragA(x + (size_t)row0 * DIN + kb * 32, DIN, lane);
  if (row0 + 16 < N) __builtin_prefetch(x + (size_t)(row0 + 16) * DIN, 0, 0);

  // q = x@Wq + bq
#pragma unroll
  for (int nb = 0; nb < 8; ++nb) {
    v8f acc = bias_bcast(bq, nb, lane);
#pragma unroll
    for (int kb = 0; kb < 4; ++kb) WMMA_BF16(acc, A[kb], pQ[(size_t)(kb * 8 + nb) * 32 + lane]);
    store_D(q + (size_t)row0 * HC + nb * 16, HC, lane, acc);
  }
  // k = x@Wk + bk
#pragma unroll
  for (int nb = 0; nb < 8; ++nb) {
    v8f acc = bias_bcast(bk, nb, lane);
#pragma unroll
    for (int kb = 0; kb < 4; ++kb) WMMA_BF16(acc, A[kb], pK[(size_t)(kb * 8 + nb) * 32 + lane]);
    store_D(k + (size_t)row0 * HC + nb * 16, HC, lane, acc);
  }
  // v = x@Wv + bv   -> staged in LDS
#pragma unroll
  for (int nb = 0; nb < 8; ++nb) {
    v8f acc = bias_bcast(bv, nb, lane);
#pragma unroll
    for (int kb = 0; kb < 4; ++kb) WMMA_BF16(acc, A[kb], pV[(size_t)(kb * 8 + nb) * 32 + lane]);
    store_D(&vtile[0][nb * 16], 132, lane, acc);
  }
  __syncthreads();
  // vcon[:,h,:] = v[:,h,:] @ Wcon_h  (K=32 -> single WMMA per 16x16 tile)
#pragma unroll
  for (int h = 0; h < 4; ++h) {
    v16bf Av = load_fragA(&vtile[0][h * 32], 132, lane);
#pragma unroll
    for (int nb = 0; nb < 2; ++nb) {
      v8f acc = zero8();
      WMMA_BF16(acc, Av, pCon[(size_t)(h * 2 + nb) * 32 + lane]);
      store_D(vcon + (size_t)row0 * HC + h * 32 + nb * 16, HC, lane, acc);
    }
  }
  // attr = x @ Wh2
#pragma unroll
  for (int nb = 0; nb < 2; ++nb) {
    v8f acc = zero8();
#pragma unroll
    for (int kb = 0; kb < 4; ++kb) WMMA_BF16(acc, A[kb], pH2[(size_t)(kb * 2 + nb) * 32 + lane]);
    store_D(attr + (size_t)row0 * C + nb * 16, C, lane, acc);
  }
}

// ---------------------------------------------------------------------------
// Edge pass A: ea2 tile (WMMA, LDS) -> logits l1=q[dst].k[src], l2=q[dst].ea2
// atomic segment max; also xh = ea@Wh1 for first E/2 edges
// ---------------------------------------------------------------------------
__global__ __launch_bounds__(32) void edge_pass_a(
    const float* __restrict__ edge_attr, const int* __restrict__ ei,
    const float* __restrict__ be2,
    const v16bf* __restrict__ pE2, const v16bf* __restrict__ pH1,
    const float* __restrict__ q, const float* __restrict__ k,
    float* __restrict__ l1o, float* __restrict__ l2o,
    unsigned* __restrict__ m1, unsigned* __restrict__ m2,
    float* __restrict__ xh) {
  __shared__ float ea2t[16][132];
  int lane = threadIdx.x;
  int e0 = blockIdx.x * 16;

  v16bf A[4];
#pragma unroll
  for (int kb = 0; kb < 4; ++kb)
    A[kb] = load_fragA(edge_attr + (size_t)e0 * DIN + kb * 32, DIN, lane);
  if (e0 + 16 < E) __builtin_prefetch(edge_attr + (size_t)(e0 + 16) * DIN, 0, 0);

#pragma unroll
  for (int nb = 0; nb < 8; ++nb) {
    v8f acc = bias_bcast(be2, nb, lane);
#pragma unroll
    for (int kb = 0; kb < 4; ++kb) WMMA_BF16(acc, A[kb], pE2[(size_t)(kb * 8 + nb) * 32 + lane]);
    store_D(&ea2t[0][nb * 16], 132, lane, acc);
  }
  if (e0 < E2) {
#pragma unroll
    for (int nb = 0; nb < 2; ++nb) {
      v8f acc = zero8();
#pragma unroll
      for (int kb = 0; kb < 4; ++kb) WMMA_BF16(acc, A[kb], pH1[(size_t)(kb * 2 + nb) * 32 + lane]);
      store_D(xh + (size_t)e0 * C + nb * 16, C, lane, acc);
    }
  }
  __syncthreads();

  int m = lane & 15;
  int e = e0 + m;
  int src = ei[e], dst = ei[E + e];
  const float* qrow = q + (size_t)dst * HC;
  const float* krow = k + (size_t)src * HC;
  int hbase = (lane >> 4) * 2;
#pragma unroll
  for (int hh = 0; hh < 2; ++hh) {
    int h = hbase + hh;
    float d1 = 0.f, d2 = 0.f;
#pragma unroll
    for (int c4 = 0; c4 < 8; ++c4) {
      float4 qv = *(const float4*)(qrow + h * 32 + c4 * 4);
      float4 kv = *(const float4*)(krow + h * 32 + c4 * 4);
      d1 += qv.x * kv.x + qv.y * kv.y + qv.z * kv.z + qv.w * kv.w;
      const float* ep = &ea2t[m][h * 32 + c4 * 4];
      d2 += qv.x * ep[0] + qv.y * ep[1] + qv.z * ep[2] + qv.w * ep[3];
    }
    d1 *= SCALE; d2 *= SCALE;
    l1o[(size_t)e * 4 + h] = d1;
    l2o[(size_t)e * 4 + h] = d2;
    atomicMax(&m1[(size_t)dst * 4 + h], enc_ord(d1));
    atomicMax(&m2[(size_t)dst * 4 + h], enc_ord(d2));
  }
}

// ---------------------------------------------------------------------------
// Edge pass B: recompute eacon = ea@M3+be3con (WMMA), exps, atomic numerators
// ---------------------------------------------------------------------------
__global__ __launch_bounds__(32) void edge_pass_b(
    const float* __restrict__ edge_attr, const int* __restrict__ ei,
    const float* __restrict__ be3con, const v16bf* __restrict__ pM3,
    const float* __restrict__ vcon,
    const float* __restrict__ l1, const float* __restrict__ l2,
    const unsigned* __restrict__ m1, const unsigned* __restrict__ m2,
    float* __restrict__ s1, float* __restrict__ s2,
    float* __restrict__ acc1, float* __restrict__ acc2) {
  __shared__ float eacont[16][132];
  __shared__ float w1[16][4], w2[16][4];
  __shared__ int ssrc[16], sdst[16];
  int lane = threadIdx.x;
  int e0 = blockIdx.x * 16;

  v16bf A[4];
#pragma unroll
  for (int kb = 0; kb < 4; ++kb)
    A[kb] = load_fragA(edge_attr + (size_t)e0 * DIN + kb * 32, DIN, lane);
  if (e0 + 16 < E) __builtin_prefetch(edge_attr + (size_t)(e0 + 16) * DIN, 0, 0);

#pragma unroll
  for (int nb = 0; nb < 8; ++nb) {
    v8f acc = bias_bcast(be3con, nb, lane);
#pragma unroll
    for (int kb = 0; kb < 4; ++kb) WMMA_BF16(acc, A[kb], pM3[(size_t)(kb * 8 + nb) * 32 + lane]);
    store_D(&eacont[0][nb * 16], 132, lane, acc);
  }

  int m = lane & 15;
  int e = e0 + m;
  int src = ei[e], dst = ei[E + e];
  if (lane < 16) { ssrc[m] = src; sdst[m] = dst; }
  int hbase = (lane >> 4) * 2;
#pragma unroll
  for (int hh = 0; hh < 2; ++hh) {
    int h = hbase + hh;
    float ex1 = expf(l1[(size_t)e * 4 + h] - dec_ord(m1[(size_t)dst * 4 + h]));
    float ex2 = expf(l2[(size_t)e * 4 + h] - dec_ord(m2[(size_t)dst * 4 + h]));
    atomicAdd(&s1[(size_t)dst * 4 + h], ex1);
    atomicAdd(&s2[(size_t)dst * 4 + h], ex2);
    w1[m][h] = ex1; w2[m][h] = ex2;
  }
  __syncthreads();

  for (int mm = 0; mm < 16; ++mm) {
    int d = sdst[mm], s = ssrc[mm];
#pragma unroll
    for (int t = 0; t < 4; ++t) {
      int col = t * 32 + lane;   // head = t
      atomicAdd(&acc1[(size_t)d * HC + col], w1[mm][t] * vcon[(size_t)s * HC + col]);
      atomicAdd(&acc2[(size_t)d * HC + col], w2[mm][t] * eacont[mm][col]);
    }
  }
}

// ---------------------------------------------------------------------------
// Node finalize: divide by softmax sums, head-reduce; partials for mean/var
// ---------------------------------------------------------------------------
__global__ __launch_bounds__(256) void node_finalize_kernel(
    const float* __restrict__ acc1, const float* __restrict__ acc2,
    const float* __restrict__ s1, const float* __restrict__ s2,
    float* __restrict__ xraw, float* __restrict__ sums) {
  __shared__ float r1[256], r2[256];
  int tid = threadIdx.x;
  size_t idx = (size_t)blockIdx.x * 256 + tid;
  int n = (int)(idx >> 5), c = (int)(idx & 31);
  float r = 0.f;
#pragma unroll
  for (int h = 0; h < 4; ++h) {
    r += acc1[(size_t)n * HC + h * 32 + c] / (s1[(size_t)n * 4 + h] + 1e-16f)
       + acc2[(size_t)n * HC + h * 32 + c] / (s2[(size_t)n * 4 + h] + 1e-16f);
  }
  xraw[idx] = r;
  r1[tid] = r; r2[tid] = r * r;
  __syncthreads();
  if (tid < 32) {
    float a = 0.f, b = 0.f;
#pragma unroll
    for (int j = 0; j < 8; ++j) { a += r1[tid + 32 * j]; b += r2[tid + 32 * j]; }
    atomicAdd(&sums[tid], a);
    atomicAdd(&sums[32 + tid], b);
  }
}

__global__ __launch_bounds__(256) void bn_gelu_kernel(
    const float* __restrict__ xraw, const float* __restrict__ sums,
    const float* __restrict__ bn_w, const float* __restrict__ bn_b,
    float* __restrict__ out) {
  size_t idx = (size_t)blockIdx.x * 256 + threadIdx.x;
  int c = (int)(idx & 31);
  float invN = 1.0f / (float)N;
  float mu = sums[c] * invN;
  float var = sums[32 + c] * invN - mu * mu;
  float xn = bn_w[c] * (xraw[idx] - mu) * rsqrtf(var + 1e-5f) + bn_b[c];
  out[idx] = 0.5f * xn * (1.0f + erff(xn * 0.70710678118654752f));
}

// ---------------------------------------------------------------------------
// Hypergraph: scatter xh into node accumulators + degree; combine; final edges
// ---------------------------------------------------------------------------
__global__ __launch_bounds__(256) void hyper_scatter_kernel(
    const float* __restrict__ xh, const int* __restrict__ ei,
    float* __restrict__ o1acc, float* __restrict__ degb) {
  size_t idx = (size_t)blockIdx.x * 256 + threadIdx.x;
  int i = (int)(idx >> 5), c = (int)(idx & 31);
  int a = ei[i], b = ei[E + i];
  float v = xh[(size_t)i * C + c];
  atomicAdd(&o1acc[(size_t)a * C + c], v);
  atomicAdd(&o1acc[(size_t)b * C + c], v);
  if (c == 0) {
    atomicAdd(&degb[a], 1.0f);
    atomicAdd(&degb[b], 1.0f);
  }
}

__global__ __launch_bounds__(256) void hyper_o1_kernel(
    const float* __restrict__ o1acc, const float* __restrict__ degb,
    const float* __restrict__ attr, float* __restrict__ o1) {
  size_t idx = (size_t)blockIdx.x * 256 + threadIdx.x;
  int n = (int)(idx >> 5);
  float dg = degb[n];
  float binv = dg > 0.f ? 1.0f / dg : 0.0f;
  o1[idx] = binv * o1acc[idx] + attr[idx];
}

__global__ __launch_bounds__(256) void hyper_out_kernel(
    const float* __restrict__ o1, const int* __restrict__ ei,
    const float* __restrict__ bh, float* __restrict__ out2) {
  size_t idx = (size_t)blockIdx.x * 256 + threadIdx.x;
  int i = (int)(idx >> 5), c = (int)(idx & 31);
  int a = ei[i], b = ei[E + i];
  out2[idx] = 0.5f * (o1[(size_t)a * C + c] + o1[(size_t)b * C + c]) + bh[c];
}

// ---------------------------------------------------------------------------
// Launch
// ---------------------------------------------------------------------------
extern "C" void kernel_launch(void* const* d_in, const int* in_sizes, int n_in,
                              void* d_out, int out_size, void* d_ws, size_t ws_size,
                              hipStream_t stream) {
  const float* x    = (const float*)d_in[0];
  const float* ea   = (const float*)d_in[1];
  const float* Wq   = (const float*)d_in[2];
  const float* bq   = (const float*)d_in[3];
  const float* Wk   = (const float*)d_in[4];
  const float* bk   = (const float*)d_in[5];
  const float* Wv   = (const float*)d_in[6];
  const float* bv   = (const float*)d_in[7];
  const float* We2  = (const float*)d_in[8];
  const float* be2  = (const float*)d_in[9];
  const float* We3  = (const float*)d_in[10];
  const float* be3  = (const float*)d_in[11];
  const float* Wcon = (const float*)d_in[12];
  const float* bnw  = (const float*)d_in[13];
  const float* bnb  = (const float*)d_in[14];
  const float* Wh1  = (const float*)d_in[15];
  const float* Wh2  = (const float*)d_in[16];
  const float* bh   = (const float*)d_in[17];
  const int*   ei   = (const int*)d_in[18];

  float* ws = (float*)d_ws;
  float* q      = ws + F_Q;
  float* k      = ws + F_K;
  float* vcon   = ws + F_VCON;
  float* attr   = ws + F_ATTR;
  float* xraw   = ws + F_XRAW;
  float* l1     = ws + F_L1;
  float* l2     = ws + F_L2;
  float* s1     = ws + F_S1;
  float* s2     = ws + F_S2;
  float* acc1   = ws + F_ACC1;
  float* acc2   = ws + F_ACC2;
  float* xh     = ws + F_XH;
  float* degb   = ws + F_DEGB;
  float* o1acc  = ws + F_O1ACC;
  float* o1     = ws + F_O1;
  float* sums   = ws + F_SUMS;
  float* M3     = ws + F_M3;
  float* be3con = ws + F_BE3CON;
  unsigned* m1  = (unsigned*)(ws + F_M1);
  unsigned* m2  = (unsigned*)(ws + F_M2);
  v16bf* pack = (v16bf*)((char*)d_ws + PACK_BYTE);
  v16bf* pQ  = pack + P_Q;
  v16bf* pK  = pack + P_K;
  v16bf* pV  = pack + P_V;
  v16bf* pE2 = pack + P_E2;
  v16bf* pM3 = pack + P_M3;
  v16bf* pCn = pack + P_CON;
  v16bf* pH1 = pack + P_H1;
  v16bf* pH2 = pack + P_H2;

  float* out_xgat  = (float*)d_out;
  float* out_edge2 = out_xgat + (size_t)N * C;

  // --- prep: composite M3 / be3con, then pack all B operands to bf16 frags ---
  build_m3_kernel<<<64, 256, 0, stream>>>(We3, Wcon, be3, M3, be3con);
  pack_b_kernel<<<32, 32, 0, stream>>>(Wq, pQ, 128);
  pack_b_kernel<<<32, 32, 0, stream>>>(Wk, pK, 128);
  pack_b_kernel<<<32, 32, 0, stream>>>(Wv, pV, 128);
  pack_b_kernel<<<32, 32, 0, stream>>>(We2, pE2, 128);
  pack_b_kernel<<<32, 32, 0, stream>>>(M3, pM3, 128);
  pack_b_kernel<<<8, 32, 0, stream>>>(Wcon, pCn, 32);
  pack_b_kernel<<<8, 32, 0, stream>>>(Wh1, pH1, 32);
  pack_b_kernel<<<8, 32, 0, stream>>>(Wh2, pH2, 32);
  init_kernel<<<(N * 128 + 255) / 256, 256, 0, stream>>>(m1, m2, s1, s2, acc1, acc2,
                                                         sums, degb, o1acc);

  // --- node GEMMs (WMMA bf16) ---
  node_gemm_kernel<<<N / 16, 32, 0, stream>>>(x, bq, bk, bv, pQ, pK, pV, pCn, pH2,
                                              q, k, vcon, attr);
  // --- edge passes ---
  edge_pass_a<<<E / 16, 32, 0, stream>>>(ea, ei, be2, pE2, pH1, q, k, l1, l2, m1, m2, xh);
  edge_pass_b<<<E / 16, 32, 0, stream>>>(ea, ei, be3con, pM3, vcon, l1, l2, m1, m2,
                                         s1, s2, acc1, acc2);
  // --- node finalize + BN + GELU ---
  node_finalize_kernel<<<(N * C) / 256, 256, 0, stream>>>(acc1, acc2, s1, s2, xraw, sums);
  bn_gelu_kernel<<<(N * C) / 256, 256, 0, stream>>>(xraw, sums, bnw, bnb, out_xgat);
  // --- hypergraph branch ---
  hyper_scatter_kernel<<<(E2 * C) / 256, 256, 0, stream>>>(xh, ei, o1acc, degb);
  hyper_o1_kernel<<<(N * C) / 256, 256, 0, stream>>>(o1acc, degb, attr, o1);
  hyper_out_kernel<<<(E2 * C) / 256, 256, 0, stream>>>(o1, ei, bh, out_edge2);
}
